// NodeNetwork_35399120454035
// MI455X (gfx1250) — matrix operand order repaired and checked by reading, not compiled, not run on hardware
//
#include <hip/hip_runtime.h>
#include <hip/hip_bf16.h>

typedef __attribute__((ext_vector_type(16))) __bf16 v16bf;
typedef __attribute__((ext_vector_type(8)))  __bf16 v8bf;
typedef __attribute__((ext_vector_type(8)))  float  v8f;

#define N_NODES  100000
#define N_EDGES  1600000
#define NODE_DIM 64
#define HIDDEN   64

// ---------------------------------------------------------------------------
// WMMA layout helpers (wave32, 16x16x32 bf16):
//  A (16xK, 16-bit): lane l holds row r = l&15. sub = l>>4.
//    element j in [0,8)  -> K = 32*c + 8*sub + j
//    element j in [8,16) -> K = 32*c + 16 + 8*sub + (j-8)
//  B (Kx16, 16-bit): lane l holds col n = l&15.
//    element j -> K = 32*c + 8*sub + j (j<8), 32*c + 16 + 8*sub + (j-8) (j>=8)
//  C/D (16x16 f32): lane l holds col n = l&15; VGPR v -> row m = v + 8*sub
// ---------------------------------------------------------------------------

__device__ __forceinline__ v16bf load_b_tile(const float* __restrict__ W,
                                             int c, int t, int sub, int ln) {
  v16bf b;
#pragma unroll
  for (int j = 0; j < 16; ++j) {
    int k = 32 * c + (j < 8 ? 8 * sub + j : 16 + 8 * sub + (j - 8));
    b[j] = (__bf16)W[k * 64 + 16 * t + ln];
  }
  return b;
}

// base points at the 32-float K-chunk of this lane's row (16B-aligned)
__device__ __forceinline__ v16bf load_a_chunk_f32(const float* __restrict__ base,
                                                  int sub) {
  const float4 x0 = *(const float4*)(base + 8 * sub);
  const float4 x1 = *(const float4*)(base + 8 * sub + 4);
  const float4 y0 = *(const float4*)(base + 16 + 8 * sub);
  const float4 y1 = *(const float4*)(base + 16 + 8 * sub + 4);
  v16bf a;
  a[0]  = (__bf16)x0.x; a[1]  = (__bf16)x0.y; a[2]  = (__bf16)x0.z; a[3]  = (__bf16)x0.w;
  a[4]  = (__bf16)x1.x; a[5]  = (__bf16)x1.y; a[6]  = (__bf16)x1.z; a[7]  = (__bf16)x1.w;
  a[8]  = (__bf16)y0.x; a[9]  = (__bf16)y0.y; a[10] = (__bf16)y0.z; a[11] = (__bf16)y0.w;
  a[12] = (__bf16)y1.x; a[13] = (__bf16)y1.y; a[14] = (__bf16)y1.z; a[15] = (__bf16)y1.w;
  return a;
}

// rowbase = LDS row of this lane's A-row (bf16, 64 elems). 16B-aligned chunks.
__device__ __forceinline__ v16bf load_a_chunk_lds(const __bf16* rowbase, int c, int sub) {
  v8bf lo = *(const v8bf*)(rowbase + 32 * c + 8 * sub);
  v8bf hi = *(const v8bf*)(rowbase + 32 * c + 16 + 8 * sub);
  return __builtin_shufflevector(lo, hi, 0, 1, 2, 3, 4, 5, 6, 7,
                                 8, 9, 10, 11, 12, 13, 14, 15);
}

__device__ __forceinline__ float leaky(float v) { return v >= 0.0f ? v : 0.1f * v; }

// ---------------------------------------------------------------------------
__global__ void zero_f4_kernel(float4* __restrict__ p, int n4) {
  int i = blockIdx.x * blockDim.x + threadIdx.x;
  if (i < n4) p[i] = make_float4(0.f, 0.f, 0.f, 0.f);
}

// ---------------------------------------------------------------------------
// Edge kernel: one wave handles 16 edges per tile iteration.
//   h   = leaky([srcFeat|edgeAttr] @ mW1 + mb1)         (16x96)@(96x64)
//   msg = (h @ mW2 + mb2) * w                            (16x64)@(64x64)
//   atomic scatter-add into agg[dst]
// ---------------------------------------------------------------------------
__global__ void __launch_bounds__(256)
gnn_edge_kernel(const float* __restrict__ nodef,
                const int*   __restrict__ eidx,
                const float* __restrict__ eattr,
                const float* __restrict__ ew,
                const float* __restrict__ mW1, const float* __restrict__ mb1,
                const float* __restrict__ mW2, const float* __restrict__ mb2,
                float* __restrict__ agg) {
  __shared__ __attribute__((aligned(16))) __bf16 hbuf[8 * 16 * 64];
  const int lane = threadIdx.x & 31;
  const int ln   = lane & 15;
  const int sub  = lane >> 4;
  __bf16* hw = hbuf + (threadIdx.x >> 5) * (16 * 64);

  const int* src_idx = eidx;
  const int* dst_idx = eidx + N_EDGES;

  // Register-resident bf16 weights in WMMA B layout (loop invariant)
  v16bf B1[3][4], B2[2][4];
#pragma unroll
  for (int c = 0; c < 3; ++c)
#pragma unroll
    for (int t = 0; t < 4; ++t) B1[c][t] = load_b_tile(mW1, c, t, sub, ln);
#pragma unroll
  for (int c = 0; c < 2; ++c)
#pragma unroll
    for (int t = 0; t < 4; ++t) B2[c][t] = load_b_tile(mW2, c, t, sub, ln);

  float b1v[4], b2v[4];
#pragma unroll
  for (int t = 0; t < 4; ++t) { b1v[t] = mb1[16 * t + ln]; b2v[t] = mb2[16 * t + ln]; }

  const int ntiles = N_EDGES / 16;
  const int nwaves = (gridDim.x * blockDim.x) >> 5;
  const int gwave  = (blockIdx.x * blockDim.x + threadIdx.x) >> 5;

  for (int tile = gwave; tile < ntiles; tile += nwaves) {
    const int e0 = tile * 16;
    const int er = e0 + ln;                 // this lane's A row (edge)
    const int s  = src_idx[er];

    // A = [srcFeat(64) | edgeAttr(32)] in bf16 WMMA-A layout, 3 K-chunks
    v16bf A0 = load_a_chunk_f32(nodef + (size_t)s  * 64,      sub);
    v16bf A1 = load_a_chunk_f32(nodef + (size_t)s  * 64 + 32, sub);
    v16bf A2 = load_a_chunk_f32(eattr + (size_t)er * 32,      sub);

    // Layer 1: 12 WMMAs
    v8f acc[4];
#pragma unroll
    for (int t = 0; t < 4; ++t) {
#pragma unroll
      for (int i = 0; i < 8; ++i) acc[t][i] = b1v[t];
      acc[t] = __builtin_amdgcn_wmma_f32_16x16x32_bf16(false, A0, false, B1[0][t],
                                                       (short)0, acc[t], false, false);
      acc[t] = __builtin_amdgcn_wmma_f32_16x16x32_bf16(false, A1, false, B1[1][t],
                                                       (short)0, acc[t], false, false);
      acc[t] = __builtin_amdgcn_wmma_f32_16x16x32_bf16(false, A2, false, B1[2][t],
                                                       (short)0, acc[t], false, false);
    }

    // LeakyReLU then re-layout D -> A through per-wave LDS scratch
#pragma unroll
    for (int t = 0; t < 4; ++t)
#pragma unroll
      for (int i = 0; i < 8; ++i)
        hw[(8 * sub + i) * 64 + 16 * t + ln] = (__bf16)leaky(acc[t][i]);
    asm volatile("s_wait_dscnt 0" ::: "memory");

    v16bf H0 = load_a_chunk_lds(hw + ln * 64, 0, sub);
    v16bf H1 = load_a_chunk_lds(hw + ln * 64, 1, sub);

    // Layer 2: 8 WMMAs
    v8f acc2[4];
#pragma unroll
    for (int t = 0; t < 4; ++t) {
#pragma unroll
      for (int i = 0; i < 8; ++i) acc2[t][i] = b2v[t];
      acc2[t] = __builtin_amdgcn_wmma_f32_16x16x32_bf16(false, H0, false, B2[0][t],
                                                        (short)0, acc2[t], false, false);
      acc2[t] = __builtin_amdgcn_wmma_f32_16x16x32_bf16(false, H1, false, B2[1][t],
                                                        (short)0, acc2[t], false, false);
    }

    // edge-weight scale + scatter-add (hits L2 atomic units; agg fits in L2)
#pragma unroll
    for (int i = 0; i < 8; ++i) {
      const int  em = e0 + 8 * sub + i;     // edge of D-row m = i + 8*sub
      const float w = ew[em];
      const int   d = dst_idx[em];
      float* dp = agg + (size_t)d * 64 + ln;
#pragma unroll
      for (int t = 0; t < 4; ++t)
        unsafeAtomicAdd(dp + 16 * t, acc2[t][i] * w);
    }
  }
}

// ---------------------------------------------------------------------------
// Node kernel: one wave handles 16 nodes per tile iteration.
//   z = [nodeFeat|agg] @ uW1 + ub1 ; LayerNorm ; leaky ; @ uW2 + ub2
// ---------------------------------------------------------------------------
__global__ void __launch_bounds__(256)
gnn_node_kernel(const float* __restrict__ nodef,
                const float* __restrict__ agg,
                const float* __restrict__ uW1, const float* __restrict__ ub1,
                const float* __restrict__ lng, const float* __restrict__ lnb,
                const float* __restrict__ uW2, const float* __restrict__ ub2,
                float* __restrict__ out) {
  __shared__ __attribute__((aligned(16))) __bf16 hbuf[8 * 16 * 64];
  const int lane = threadIdx.x & 31;
  const int ln   = lane & 15;
  const int sub  = lane >> 4;
  __bf16* hw = hbuf + (threadIdx.x >> 5) * (16 * 64);

  v16bf B1[4][4], B2[2][4];
#pragma unroll
  for (int c = 0; c < 4; ++c)
#pragma unroll
    for (int t = 0; t < 4; ++t) B1[c][t] = load_b_tile(uW1, c, t, sub, ln);
#pragma unroll
  for (int c = 0; c < 2; ++c)
#pragma unroll
    for (int t = 0; t < 4; ++t) B2[c][t] = load_b_tile(uW2, c, t, sub, ln);

  float b1v[4], b2v[4], g4[4], bb4[4];
#pragma unroll
  for (int t = 0; t < 4; ++t) {
    b1v[t] = ub1[16 * t + ln]; b2v[t] = ub2[16 * t + ln];
    g4[t]  = lng[16 * t + ln]; bb4[t] = lnb[16 * t + ln];
  }

  const int ntiles = N_NODES / 16;
  const int nwaves = (gridDim.x * blockDim.x) >> 5;
  const int gwave  = (blockIdx.x * blockDim.x + threadIdx.x) >> 5;

  for (int tile = gwave; tile < ntiles; tile += nwaves) {
    const int r = tile * 16 + ln;           // this lane's A row (node)

    v16bf A0 = load_a_chunk_f32(nodef + (size_t)r * 64,      sub);
    v16bf A1 = load_a_chunk_f32(nodef + (size_t)r * 64 + 32, sub);
    v16bf A2 = load_a_chunk_f32(agg   + (size_t)r * 64,      sub);
    v16bf A3 = load_a_chunk_f32(agg   + (size_t)r * 64 + 32, sub);

    // Layer 1: 16 WMMAs
    v8f acc[4];
#pragma unroll
    for (int t = 0; t < 4; ++t) {
#pragma unroll
      for (int i = 0; i < 8; ++i) acc[t][i] = b1v[t];
      acc[t] = __builtin_amdgcn_wmma_f32_16x16x32_bf16(false, A0, false, B1[0][t],
                                                       (short)0, acc[t], false, false);
      acc[t] = __builtin_amdgcn_wmma_f32_16x16x32_bf16(false, A1, false, B1[1][t],
                                                       (short)0, acc[t], false, false);
      acc[t] = __builtin_amdgcn_wmma_f32_16x16x32_bf16(false, A2, false, B1[2][t],
                                                       (short)0, acc[t], false, false);
      acc[t] = __builtin_amdgcn_wmma_f32_16x16x32_bf16(false, A3, false, B1[3][t],
                                                       (short)0, acc[t], false, false);
    }

    // LayerNorm over 64 cols of each row m = i + 8*sub.
    // Each 16-lane half owns all 64 columns of its 8 rows -> shfl_xor reduce.
#pragma unroll
    for (int i = 0; i < 8; ++i) {
      float s = acc[0][i] + acc[1][i] + acc[2][i] + acc[3][i];
      float q = acc[0][i] * acc[0][i] + acc[1][i] * acc[1][i] +
                acc[2][i] * acc[2][i] + acc[3][i] * acc[3][i];
#pragma unroll
      for (int m = 1; m < 16; m <<= 1) {
        s += __shfl_xor(s, m, 32);
        q += __shfl_xor(q, m, 32);
      }
      const float mu  = s * 0.015625f;           // /64
      const float var = q * 0.015625f - mu * mu;
      const float rs  = rsqrtf(var + 1e-5f);
#pragma unroll
      for (int t = 0; t < 4; ++t) {
        float z = (acc[t][i] - mu) * rs * g4[t] + bb4[t];
        hw[(8 * sub + i) * 64 + 16 * t + ln] = (__bf16)leaky(z);
      }
    }
    asm volatile("s_wait_dscnt 0" ::: "memory");

    v16bf H0 = load_a_chunk_lds(hw + ln * 64, 0, sub);
    v16bf H1 = load_a_chunk_lds(hw + ln * 64, 1, sub);

    // Layer 2: 8 WMMAs
    v8f acc2[4];
#pragma unroll
    for (int t = 0; t < 4; ++t) {
#pragma unroll
      for (int i = 0; i < 8; ++i) acc2[t][i] = b2v[t];
      acc2[t] = __builtin_amdgcn_wmma_f32_16x16x32_bf16(false, H0, false, B2[0][t],
                                                        (short)0, acc2[t], false, false);
      acc2[t] = __builtin_amdgcn_wmma_f32_16x16x32_bf16(false, H1, false, B2[1][t],
                                                        (short)0, acc2[t], false, false);
    }

    // Coalesced f32 store (16 consecutive floats per half-wave per (t,i))
#pragma unroll
    for (int i = 0; i < 8; ++i) {
      float* rowp = out + (size_t)(tile * 16 + 8 * sub + i) * 64 + ln;
#pragma unroll
      for (int t = 0; t < 4; ++t) rowp[16 * t] = acc2[t][i];
    }
  }
}

// ---------------------------------------------------------------------------
extern "C" void kernel_launch(void* const* d_in, const int* in_sizes, int n_in,
                              void* d_out, int out_size, void* d_ws, size_t ws_size,
                              hipStream_t stream) {
  (void)in_sizes; (void)n_in; (void)out_size;
  const float* nodef = (const float*)d_in[0];
  const int*   eidx  = (const int*)  d_in[1];
  const float* eattr = (const float*)d_in[2];
  const float* ew    = (const float*)d_in[3];
  const float* mW1   = (const float*)d_in[4];
  const float* mb1   = (const float*)d_in[5];
  const float* mW2   = (const float*)d_in[6];
  const float* mb2   = (const float*)d_in[7];
  const float* uW1   = (const float*)d_in[8];
  const float* ub1   = (const float*)d_in[9];
  const float* lng   = (const float*)d_in[10];
  const float* lnb   = (const float*)d_in[11];
  const float* uW2   = (const float*)d_in[12];
  const float* ub2   = (const float*)d_in[13];
  float* out = (float*)d_out;

  // aggregated buffer: d_ws if big enough, else in-place via d_out
  // (in-place is safe: each node tile reads its agg rows before writing them)
  const size_t need = (size_t)N_NODES * NODE_DIM * sizeof(float);
  float* agg = (ws_size >= need) ? (float*)d_ws : out;

  const int n4 = N_NODES * NODE_DIM / 4;
  zero_f4_kernel<<<(n4 + 255) / 256, 256, 0, stream>>>((float4*)agg, n4);

  gnn_edge_kernel<<<2048, 256, 0, stream>>>(nodef, eidx, eattr, ew,
                                            mW1, mb1, mW2, mb2, agg);

  gnn_node_kernel<<<512, 256, 0, stream>>>(nodef, agg, uW1, ub1,
                                           lng, lnb, uW2, ub2, out);
}
